// CrossAttention_63857573757329
// MI455X (gfx1250) — matrix operand order repaired and checked by reading, not compiled, not run on hardware
//
#include <hip/hip_runtime.h>

typedef __attribute__((ext_vector_type(8)))  float          v8f;
typedef __attribute__((ext_vector_type(16))) __bf16         v16bf;
typedef __attribute__((ext_vector_type(16))) unsigned short us16;
typedef __attribute__((ext_vector_type(8)))  unsigned short us8;

__device__ __forceinline__ unsigned short bf16_rn(float f) {
    union { float f; unsigned int u; } v; v.f = f;
    unsigned int u = v.u;
    u += 0x7fffu + ((u >> 16) & 1u);
    return (unsigned short)(u >> 16);
}

union BC16 { us16 u; v16bf b; };
__device__ __forceinline__ v16bf as_bf(us16 u) { BC16 x; x.u = u; return x.b; }

__device__ __forceinline__ float fexp2(float x) { return __builtin_amdgcn_exp2f(x); }

#define L_SEQ   1024
#define DIMV    640
#define INNERV  256
#define NKB     8        // 256/32 K-blocks over INNER
#define NKB_W   20       // 640/32 K-blocks over DIM

// ---------------------------------------------------------------------------
// K0: pack Wq/Wk [DIM, INNER] fp32 -> bf16 B-fragment layout:
//     Wp[((kb*16 + nt)*32 + lane)*16 + j] = W[kb*32 + (lane>>4)*8 + (j&7) + (j>>3)*16][nt*16 + (lane&15)]
// ---------------------------------------------------------------------------
__global__ __launch_bounds__(256)
void pack_w(const float* __restrict__ Wq, const float* __restrict__ Wk,
            unsigned short* __restrict__ Wqp, unsigned short* __restrict__ Wkp) {
    int idx = blockIdx.x * 256 + threadIdx.x;
    if (idx >= NKB_W * 16 * 32 * 16) return;
    int j    = idx & 15;
    int lane = (idx >> 4) & 31;
    int nt   = (idx >> 9) & 15;
    int kb   = idx >> 13;
    int k = kb * 32 + (lane >> 4) * 8 + (j & 7) + (j >> 3) * 16;
    int n = nt * 16 + (lane & 15);
    Wqp[idx] = bf16_rn(Wq[(size_t)k * INNERV + n]);
    Wkp[idx] = bf16_rn(Wk[(size_t)k * INNERV + n]);
}

// ---------------------------------------------------------------------------
// K1: projections Q = X@Wq + bq, K = X@Wk + bk  (bf16 out, row-major [L][256])
// grid: (64 row-tiles, 2 seqs, 64 batch), 256 threads (8 waves x 4 n-tiles)
// ---------------------------------------------------------------------------
__global__ __launch_bounds__(256)
void proj_kernel(const float* __restrict__ a_pad, const float* __restrict__ b_pad,
                 const unsigned short* __restrict__ Wqp, const unsigned short* __restrict__ Wkp,
                 const float* __restrict__ bq, const float* __restrict__ bk,
                 unsigned short* __restrict__ QA, unsigned short* __restrict__ KA,
                 unsigned short* __restrict__ QB, unsigned short* __restrict__ KB) {
    const int rt = blockIdx.x, seq = blockIdx.y, b = blockIdx.z;
    const float* X = (seq == 0 ? a_pad : b_pad) + ((size_t)(b * L_SEQ + rt * 16)) * DIMV;
    unsigned short* Qo = (seq == 0 ? QA : QB) + ((size_t)(b * L_SEQ + rt * 16)) * INNERV;
    unsigned short* Ko = (seq == 0 ? KA : KB) + ((size_t)(b * L_SEQ + rt * 16)) * INNERV;

    __shared__ __align__(16) unsigned short xs[16 * DIMV];
    const int t = threadIdx.x;
    for (int i = t; i < 16 * DIMV; i += 256) xs[i] = bf16_rn(X[i]);
    __syncthreads();

    const int wave = t >> 5, lane = t & 31, lmod = lane & 15, lhalf = lane >> 4;
    const int nt0 = wave * 2;

    v8f aq0, aq1, ak0, ak1;
#pragma unroll
    for (int r = 0; r < 8; ++r) { aq0[r] = 0.f; aq1[r] = 0.f; ak0[r] = 0.f; ak1[r] = 0.f; }

    for (int kb = 0; kb < NKB_W; ++kb) {
        // A fragment (16x32 bf16) from LDS per ISA layout
        const int c0 = lmod * DIMV + kb * 32 + lhalf * 8;
        union { us16 v; us8 h[2]; } A;
        A.h[0] = *(const us8*)&xs[c0];
        A.h[1] = *(const us8*)&xs[c0 + 16];
        const v16bf af = as_bf(A.v);

        const int base = kb * 8192 + lane * 16;   // ((kb*16 + nt)*32 + lane)*16
        const us16 w0 = *(const us16*)&Wqp[base + (nt0    ) * 512];
        const us16 w1 = *(const us16*)&Wqp[base + (nt0 + 1) * 512];
        const us16 w2 = *(const us16*)&Wkp[base + (nt0    ) * 512];
        const us16 w3 = *(const us16*)&Wkp[base + (nt0 + 1) * 512];

        aq0 = __builtin_amdgcn_wmma_f32_16x16x32_bf16(false, af, false, as_bf(w0), (short)0, aq0, false, false);
        aq1 = __builtin_amdgcn_wmma_f32_16x16x32_bf16(false, af, false, as_bf(w1), (short)0, aq1, false, false);
        ak0 = __builtin_amdgcn_wmma_f32_16x16x32_bf16(false, af, false, as_bf(w2), (short)0, ak0, false, false);
        ak1 = __builtin_amdgcn_wmma_f32_16x16x32_bf16(false, af, false, as_bf(w3), (short)0, ak1, false, false);
    }

    const int col0 = nt0 * 16 + lmod;
    const float bq0 = bq[col0], bq1 = bq[col0 + 16];
    const float bk0 = bk[col0], bk1 = bk[col0 + 16];
#pragma unroll
    for (int r = 0; r < 8; ++r) {
        const int row = r + 8 * lhalf;
        Qo[row * INNERV + col0     ] = bf16_rn(aq0[r] + bq0);
        Qo[row * INNERV + col0 + 16] = bf16_rn(aq1[r] + bq1);
        Ko[row * INNERV + col0     ] = bf16_rn(ak0[r] + bk0);
        Ko[row * INNERV + col0 + 16] = bf16_rn(ak1[r] + bk1);
    }
}

// ---------------------------------------------------------------------------
// K2: attention column sums c_k = sum over valid queries of softmax(QK^T/16)
// grid: (16 query-blocks of 64, 2 dirs, 64 batch); 8 waves = 4 q-tiles x 2 key halves
// ---------------------------------------------------------------------------
__global__ __launch_bounds__(256)
void attn_kernel(const unsigned short* __restrict__ QA, const unsigned short* __restrict__ KA,
                 const unsigned short* __restrict__ QB, const unsigned short* __restrict__ KB,
                 const int* __restrict__ len_a, const int* __restrict__ len_b,
                 float* __restrict__ c_glob) {
    const int qblk = blockIdx.x, dir = blockIdx.y, b = blockIdx.z;
    const unsigned short* Q = (dir == 0 ? QA : QB) + (size_t)b * L_SEQ * INNERV;
    const unsigned short* K = (dir == 0 ? KB : KA) + (size_t)b * L_SEQ * INNERV;
    const int lenq = (dir == 0 ? len_a[b] : len_b[b]);
    const int lenk = (dir == 0 ? len_b[b] : len_a[b]);

    __shared__ float c_sm[L_SEQ];
    __shared__ float sm_m[2][4][16];
    __shared__ float sm_l[2][4][16];

    const int t = threadIdx.x;
    for (int i = t; i < L_SEQ; i += 256) c_sm[i] = 0.f;

    const int wave = t >> 5, lane = t & 31, lmod = lane & 15, lhalf = lane >> 4;
    const int qt = wave & 3, half = wave >> 2;
    const int qrow0 = qblk * 64 + qt * 16;
    const float SC = 0.09016994375f;  // log2(e) / sqrt(256)

    // hoist Q A-fragments (this wave's 16 query rows, all of INNER)
    v16bf qf[NKB];
    {
        const unsigned short* Qr = Q + (size_t)(qrow0 + lmod) * INNERV;
#pragma unroll
        for (int kb = 0; kb < NKB; ++kb) {
            union { us16 v; us8 h[2]; } A;
            A.h[0] = *(const us8*)&Qr[kb * 32 + lhalf * 16];
            A.h[1] = *(const us8*)&Qr[kb * 32 + lhalf * 16 + 8];
            qf[kb] = as_bf(A.v);
        }
    }

    float m[8], l[8];
#pragma unroll
    for (int r = 0; r < 8; ++r) { m[r] = -3.0e38f; l[r] = 0.f; }

    const int kt0 = half * 32, kt1 = kt0 + 32;

    // ---- pass 1: per-lane online softmax stats over this key half ----
    for (int kt = kt0; kt < kt1; ++kt) {
        v8f s;
#pragma unroll
        for (int r = 0; r < 8; ++r) s[r] = 0.f;
        const unsigned short* Kr = K + (size_t)(kt * 16 + lmod) * INNERV + lhalf * 16;
#pragma unroll
        for (int kb = 0; kb < NKB; ++kb) {
            const us16 w = *(const us16*)&Kr[kb * 32];
            s = __builtin_amdgcn_wmma_f32_16x16x32_bf16(false, qf[kb], false, as_bf(w), (short)0, s, false, false);
        }
        const bool kval = (kt * 16 + lmod) < lenk;
#pragma unroll
        for (int r = 0; r < 8; ++r) {
            const float tm = kval ? s[r] * SC : -3.0e38f;
            const float mn = fmaxf(m[r], tm);
            l[r] = l[r] * fexp2(m[r] - mn) + fexp2(tm - mn);
            m[r] = mn;
        }
    }

    // merge stats across the 16 lanes of each half (butterfly)
#pragma unroll
    for (int r = 0; r < 8; ++r) {
#pragma unroll
        for (int off = 1; off < 16; off <<= 1) {
            const float mo = __shfl_xor(m[r], off, 16);
            const float lo = __shfl_xor(l[r], off, 16);
            const float mn = fmaxf(m[r], mo);
            l[r] = l[r] * fexp2(m[r] - mn) + lo * fexp2(mo - mn);
            m[r] = mn;
        }
    }
    if (lmod == 0) {
#pragma unroll
        for (int r = 0; r < 8; ++r) {
            sm_m[half][qt][8 * lhalf + r] = m[r];
            sm_l[half][qt][8 * lhalf + r] = l[r];
        }
    }
    __syncthreads();

    // merge the two key halves; fold query-validity weight and 1/l
    float mf[8], invw[8];
#pragma unroll
    for (int r = 0; r < 8; ++r) {
        const int row = 8 * lhalf + r;
        const float m0 = sm_m[0][qt][row], m1 = sm_m[1][qt][row];
        const float l0 = sm_l[0][qt][row], l1 = sm_l[1][qt][row];
        const float mm = fmaxf(m0, m1);
        const float ll = l0 * fexp2(m0 - mm) + l1 * fexp2(m1 - mm);
        mf[r] = mm;
        const float w = ((qrow0 + row) < lenq) ? 1.f : 0.f;
        invw[r] = w / ll;
    }

    // ---- pass 2: recompute scores, accumulate column sums ----
    for (int kt = kt0; kt < kt1; ++kt) {
        v8f s;
#pragma unroll
        for (int r = 0; r < 8; ++r) s[r] = 0.f;
        const unsigned short* Kr = K + (size_t)(kt * 16 + lmod) * INNERV + lhalf * 16;
#pragma unroll
        for (int kb = 0; kb < NKB; ++kb) {
            const us16 w = *(const us16*)&Kr[kb * 32];
            s = __builtin_amdgcn_wmma_f32_16x16x32_bf16(false, qf[kb], false, as_bf(w), (short)0, s, false, false);
        }
        const bool kval = (kt * 16 + lmod) < lenk;
        float part = 0.f;
#pragma unroll
        for (int r = 0; r < 8; ++r) {
            const float tm = kval ? s[r] * SC : -3.0e38f;
            part += fexp2(tm - mf[r]) * invw[r];
        }
        part += __shfl_xor(part, 16, 32);
        if (lane < 16) atomicAdd(&c_sm[kt * 16 + lane], part);
    }
    __syncthreads();

    float* cg = c_glob + (size_t)(b * 2 + dir) * L_SEQ;
    for (int i = t; i < L_SEQ; i += 256) atomicAdd(&cg[i], c_sm[i]);
}

// ---------------------------------------------------------------------------
// K3: d = c^T @ X_values   (X = b_pad for dir0, a_pad for dir1)
// ---------------------------------------------------------------------------
__global__ __launch_bounds__(256)
void dvec_kernel(const float* __restrict__ a_pad, const float* __restrict__ b_pad,
                 const float* __restrict__ c_glob, float* __restrict__ dvec) {
    const int dir = blockIdx.x & 1, b = blockIdx.x >> 1;
    const float* X = (dir == 0 ? b_pad : a_pad) + (size_t)b * L_SEQ * DIMV;
    const float* cg = c_glob + (size_t)(b * 2 + dir) * L_SEQ;

    __shared__ float cs[L_SEQ];
    const int t = threadIdx.x;
    for (int i = t; i < L_SEQ; i += 256) cs[i] = cg[i];
    __syncthreads();

    const int d0 = t, d1 = t + 256, d2 = t + 512;
    float acc0 = 0.f, acc1 = 0.f, acc2 = 0.f;
#pragma unroll 4
    for (int k = 0; k < L_SEQ; ++k) {
        const float ck = cs[k];
        const float* xr = X + (size_t)k * DIMV;
        acc0 += ck * xr[d0];
        acc1 += ck * xr[d1];
        if (d2 < DIMV) acc2 += ck * xr[d2];
    }
    float* dv = dvec + (size_t)(b * 2 + dir) * DIMV;
    dv[d0] = acc0;
    dv[d1] = acc1;
    if (d2 < DIMV) dv[d2] = acc2;
}

// ---------------------------------------------------------------------------
// K4: emb = d @ Wv / len + bv   -> d_out (emb_a then emb_b)
// ---------------------------------------------------------------------------
__global__ __launch_bounds__(256)
void emb_kernel(const float* __restrict__ dvec, const float* __restrict__ Wv,
                const float* __restrict__ bv, const int* __restrict__ len_a,
                const int* __restrict__ len_b, float* __restrict__ out) {
    const int dir = blockIdx.x & 1, b = blockIdx.x >> 1;
    __shared__ float ds[DIMV];
    const float* dv = dvec + (size_t)(b * 2 + dir) * DIMV;
    const int t = threadIdx.x;
    for (int i = t; i < DIMV; i += 256) ds[i] = dv[i];
    __syncthreads();

    const float invlen = 1.0f / (float)(dir == 0 ? len_a[b] : len_b[b]);
    float a0 = 0.f, a1 = 0.f, a2 = 0.f, a3 = 0.f;
    for (int k = 0; k < DIMV; ++k) {
        const float dk = ds[k];
        const float* wr = Wv + (size_t)k * 1024 + t;
        a0 += dk * wr[0];
        a1 += dk * wr[256];
        a2 += dk * wr[512];
        a3 += dk * wr[768];
    }
    float* o = out + (size_t)dir * 65536 + (size_t)b * 1024;
    o[t      ] = a0 * invlen + bv[t      ];
    o[t + 256] = a1 * invlen + bv[t + 256];
    o[t + 512] = a2 * invlen + bv[t + 512];
    o[t + 768] = a3 * invlen + bv[t + 768];
}

// ---------------------------------------------------------------------------
extern "C" void kernel_launch(void* const* d_in, const int* in_sizes, int n_in,
                              void* d_out, int out_size, void* d_ws, size_t ws_size,
                              hipStream_t stream) {
    (void)in_sizes; (void)n_in; (void)out_size; (void)ws_size;
    const float* a_pad = (const float*)d_in[0];
    const float* b_pad = (const float*)d_in[1];
    const int*   len_a = (const int*)d_in[2];
    const int*   len_b = (const int*)d_in[3];
    const float* Wq    = (const float*)d_in[4];
    const float* bq    = (const float*)d_in[5];
    const float* Wk    = (const float*)d_in[6];
    const float* bk    = (const float*)d_in[7];
    const float* Wv    = (const float*)d_in[8];
    const float* bv    = (const float*)d_in[9];
    float* out = (float*)d_out;

    char* ws = (char*)d_ws;
    const size_t SQ = (size_t)64 * L_SEQ * INNERV * sizeof(unsigned short); // 32 MiB
    unsigned short* QA  = (unsigned short*)(ws);
    unsigned short* KA  = (unsigned short*)(ws + SQ);
    unsigned short* QB  = (unsigned short*)(ws + 2 * SQ);
    unsigned short* KB  = (unsigned short*)(ws + 3 * SQ);
    unsigned short* Wqp = (unsigned short*)(ws + 4 * SQ);
    unsigned short* Wkp = (unsigned short*)(ws + 4 * SQ + 327680);
    float* cbuf = (float*)(ws + 4 * SQ + 2 * 327680);
    float* dbuf = (float*)(ws + 4 * SQ + 2 * 327680 + 524288);

    hipMemsetAsync(cbuf, 0, 64 * 2 * L_SEQ * sizeof(float), stream);
    pack_w<<<640, 256, 0, stream>>>(Wq, Wk, Wqp, Wkp);
    proj_kernel<<<dim3(64, 2, 64), 256, 0, stream>>>(a_pad, b_pad, Wqp, Wkp, bq, bk, QA, KA, QB, KB);
    attn_kernel<<<dim3(16, 2, 64), 256, 0, stream>>>(QA, KA, QB, KB, len_a, len_b, cbuf);
    dvec_kernel<<<128, 256, 0, stream>>>(a_pad, b_pad, cbuf, dbuf);
    emb_kernel<<<128, 256, 0, stream>>>(dbuf, Wv, bv, len_a, len_b, out);
}
